// VariableInteraction_59906203844675
// MI455X (gfx1250) — matrix operand order repaired and checked by reading, not compiled, not run on hardware
//
#include <hip/hip_runtime.h>
#include <hip/hip_bf16.h>

// Problem constants (match reference)
#define BB 64
#define VV 256
#define DD 256
#define NN 65536
#define QQ 64

typedef __attribute__((ext_vector_type(16))) _Float16 v16h;
typedef __attribute__((ext_vector_type(8)))  _Float16 v8h;
typedef __attribute__((ext_vector_type(8)))  float    v8f;

// ---------------------------------------------------------------------------
// 1) Scatter-add density features per variable bin (f32 global atomics)
// ---------------------------------------------------------------------------
__global__ void scatter_kernel(const float* __restrict__ df, const int* __restrict__ vi,
                               const float* __restrict__ mask,
                               float* __restrict__ vdens, float* __restrict__ vcnt) {
    long t = (long)blockIdx.x * blockDim.x + threadIdx.x;
    if (t >= (long)BB * NN) return;
    int b = (int)(t / NN);
    int idx = vi[t];
    float m = mask[t];
    long o = (long)b * VV + idx;
    atomicAdd(&vdens[o * 3 + 0], df[t * 3 + 0] * m);
    atomicAdd(&vdens[o * 3 + 1], df[t * 3 + 1] * m);
    atomicAdd(&vdens[o * 3 + 2], df[t * 3 + 2] * m);
    atomicAdd(&vcnt[o], m);
}

// ---------------------------------------------------------------------------
// 2) Density gate MLP: Linear(3,16) -> GELU(exact) -> Linear(16,1) -> sigmoid
//    Stores the fused multiplier (1 + 0.5*sigmoid) applied to score columns.
// ---------------------------------------------------------------------------
__global__ void gate_kernel(const float* __restrict__ vdens, const float* __restrict__ vcnt,
                            const float* __restrict__ g1w, const float* __restrict__ g1b,
                            const float* __restrict__ g2w, const float* __restrict__ g2b,
                            float* __restrict__ gmul) {
    int t = blockIdx.x * blockDim.x + threadIdx.x;
    if (t >= BB * VV) return;
    float cnt = fmaxf(vcnt[t], 1.0f);
    float d0 = vdens[t * 3 + 0] / cnt;
    float d1 = vdens[t * 3 + 1] / cnt;
    float d2 = vdens[t * 3 + 2] / cnt;
    float acc = g2b[0];
#pragma unroll
    for (int j = 0; j < 16; ++j) {
        float h = g1b[j] + g1w[j * 3 + 0] * d0 + g1w[j * 3 + 1] * d1 + g1w[j * 3 + 2] * d2;
        h = 0.5f * h * (1.0f + erff(h * 0.70710678118f));   // exact GELU
        acc += g2w[j] * h;
    }
    float s = 1.0f / (1.0f + expf(-acc));
    gmul[t] = 1.0f + 0.5f * s;
}

// ---------------------------------------------------------------------------
// 3) Build f16 weights: quaternion Hamilton-block Wq/Wk, plus f16 casts of
//    v_w / o_w.  blockIdx.y selects the task.
// ---------------------------------------------------------------------------
__global__ void prep_weights_kernel(const float* __restrict__ qr, const float* __restrict__ qi,
                                    const float* __restrict__ qj, const float* __restrict__ qk,
                                    const float* __restrict__ kr, const float* __restrict__ ki,
                                    const float* __restrict__ kj, const float* __restrict__ kk,
                                    const float* __restrict__ vw, const float* __restrict__ ow,
                                    _Float16* __restrict__ Wq, _Float16* __restrict__ Wk,
                                    _Float16* __restrict__ Wv, _Float16* __restrict__ Wo) {
    int idx = blockIdx.x * blockDim.x + threadIdx.x;     // 0 .. D*D-1
    int task = blockIdx.y;
    if (idx >= DD * DD) return;
    if (task >= 2) {
        const float* s = (task == 2) ? vw : ow;
        _Float16* d = (task == 2) ? Wv : Wo;
        d[idx] = (_Float16)s[idx];
        return;
    }
    int o  = idx >> 8;          // output row (D)
    int di = idx & (DD - 1);    // input col  (D)
    int rb = o >> 6, rr = o & (QQ - 1);
    int cb = di >> 6, cc = di & (QQ - 1);
    // Hamilton block matrix: rows [r -i -j -k; i r -k j; j k r -i; k -j i r]
    const int  csel[4][4] = {{0,1,2,3},{1,0,3,2},{2,3,0,1},{3,2,1,0}};
    const float sgn[4][4] = {{1.f,-1.f,-1.f,-1.f},{1.f,1.f,-1.f,1.f},
                             {1.f,1.f,1.f,-1.f},{1.f,-1.f,1.f,1.f}};
    const float* comps[4];
    if (task == 0) { comps[0]=qr; comps[1]=qi; comps[2]=qj; comps[3]=qk; }
    else           { comps[0]=kr; comps[1]=ki; comps[2]=kj; comps[3]=kk; }
    float val = sgn[rb][cb] * comps[csel[rb][cb]][rr * QQ + cc];
    (task == 0 ? Wq : Wk)[idx] = (_Float16)val;
}

// ---------------------------------------------------------------------------
// 4) f32 -> f16 conversion (variable_he)
// ---------------------------------------------------------------------------
__global__ void cvt_f16_kernel(const float* __restrict__ src, _Float16* __restrict__ dst, long n) {
    long t = (long)blockIdx.x * blockDim.x + threadIdx.x;
    if (t < n) dst[t] = (_Float16)src[t];
}

// ---------------------------------------------------------------------------
// 5) Batched WMMA GEMM:  C[b] = A[b] (MxK, row-major f16)  x  Bw[b]^T
//    where Bw is stored row-major [N,K] (i.e. exactly W for X @ W^T).
//    One wave32 computes a 32x32 output block = 2x2 register tiles with
//    4 v_wmma_f32_16x16x32_f16 accumulators; per k-step it loads 2 A
//    fragments + 2 B fragments and issues 4 WMMAs (2x the L2 arithmetic
//    intensity of a 1x1-tile wave).
//    MODE 0: store f16 row-major; MODE 1: store f16 transposed [N,M];
//    MODE 2: store f32 row-major.  Epilogue: c*scale*gmul[col] + bias[col].
// ---------------------------------------------------------------------------
template <int MODE>
__global__ void wmma_gemm_kernel(const _Float16* __restrict__ A, long sA, int lda,
                                 const _Float16* __restrict__ Bw, long sB, int ldb,
                                 void* __restrict__ C, long sC, int ldc,
                                 const float* __restrict__ bias,
                                 const float* __restrict__ gmul, int sG,
                                 float scale, int M, int N, int K, int batches) {
    int wave = (int)((blockIdx.x * (long)blockDim.x + threadIdx.x) >> 5);
    int lane = threadIdx.x & 31;
    int tilesN = N >> 5, tilesM = M >> 5;       // 32x32 macro tiles per wave
    int tpb = tilesM * tilesN;
    if (wave >= batches * tpb) return;          // wave-uniform: EXEC stays all-1s
    int b  = wave / tpb;
    int r  = wave - b * tpb;
    int tm = r / tilesN;
    int tn = r - tm * tilesN;
    int m0 = tm << 5, n0 = tn << 5;

    int row = lane & 15;      // A-row / B-column within 16x16 tile
    int hi  = lane >> 4;      // lane-group select (ISA A/B K striping)

    const _Float16* arow0 = A  + (long)b * sA + (long)(m0 + row) * lda;
    const _Float16* arow1 = arow0 + (long)16 * lda;
    const _Float16* brow0 = Bw + (long)b * sB + (long)(n0 + row) * ldb;
    const _Float16* brow1 = brow0 + (long)16 * ldb;

    v8f c00 = {}, c01 = {}, c10 = {}, c11 = {};
    for (int k0 = 0; k0 < K; k0 += 32) {
        // A 16x32 f16: lanes 0-15 hold K {0..7,16..23}, lanes 16-31 K {8..15,24..31}
        v8h a0lo = *(const v8h*)(arow0 + k0 + hi * 8);
        v8h a0hi = *(const v8h*)(arow0 + k0 + hi * 8 + 16);
        v8h a1lo = *(const v8h*)(arow1 + k0 + hi * 8);
        v8h a1hi = *(const v8h*)(arow1 + k0 + hi * 8 + 16);
        v16h a0, a1;
#pragma unroll
        for (int t = 0; t < 8; ++t) {
            a0[t] = a0lo[t]; a0[t + 8] = a0hi[t];
            a1[t] = a1lo[t]; a1[t + 8] = a1hi[t];
        }
        // B 32x16 f16 (as W rows): lanes 0-15 K {0..7,16..23}, lanes 16-31 K {8..15,24..31}
        v8h b0lo = *(const v8h*)(brow0 + k0 + hi * 8);
        v8h b0hi = *(const v8h*)(brow0 + k0 + hi * 8 + 16);
        v8h b1lo = *(const v8h*)(brow1 + k0 + hi * 8);
        v8h b1hi = *(const v8h*)(brow1 + k0 + hi * 8 + 16);
        v16h b0, b1;
#pragma unroll
        for (int t = 0; t < 8; ++t) {
            b0[t] = b0lo[t]; b0[t + 8] = b0hi[t];
            b1[t] = b1lo[t]; b1[t + 8] = b1hi[t];
        }
        c00 = __builtin_amdgcn_wmma_f32_16x16x32_f16(false, a0, false, b0, (short)0, c00, false, false);
        c01 = __builtin_amdgcn_wmma_f32_16x16x32_f16(false, a0, false, b1, (short)0, c01, false, false);
        c10 = __builtin_amdgcn_wmma_f32_16x16x32_f16(false, a1, false, b0, (short)0, c10, false, false);
        c11 = __builtin_amdgcn_wmma_f32_16x16x32_f16(false, a1, false, b1, (short)0, c11, false, false);
    }

    // Epilogue + store: C layout per tile: VGPR j -> M = j (+8 for hi lanes), N = row
    const v8f* accs[4] = {&c00, &c01, &c10, &c11};
#pragma unroll
    for (int tt = 0; tt < 4; ++tt) {
        int tmi = tt >> 1, tni = tt & 1;
        int ocol = n0 + tni * 16 + row;
        float bval = bias ? bias[ocol] : 0.0f;
        float g    = gmul ? gmul[(long)b * sG + ocol] : 1.0f;
        const v8f& acc = *accs[tt];
#pragma unroll
        for (int j = 0; j < 8; ++j) {
            int orow = m0 + tmi * 16 + j + hi * 8;
            float v = acc[j] * scale * g + bval;
            if (MODE == 0)
                ((_Float16*)C)[(long)b * sC + (long)orow * ldc + ocol] = (_Float16)v;
            else if (MODE == 1)
                ((_Float16*)C)[(long)b * sC + (long)ocol * ldc + orow] = (_Float16)v;
            else
                ((float*)C)[(long)b * sC + (long)orow * ldc + ocol] = v;
        }
    }
}

// ---------------------------------------------------------------------------
// 6) Row softmax over V=256 scores -> f16 attention weights
// ---------------------------------------------------------------------------
__global__ void softmax_kernel(const float* __restrict__ scores, _Float16* __restrict__ w16) {
    __shared__ float red[VV];
    long base = (long)blockIdx.x * VV;
    int tid = threadIdx.x;
    float x = scores[base + tid];
    red[tid] = x; __syncthreads();
    for (int s = VV / 2; s > 0; s >>= 1) {
        if (tid < s) red[tid] = fmaxf(red[tid], red[tid + s]);
        __syncthreads();
    }
    float mx = red[0]; __syncthreads();
    float e = expf(x - mx);
    red[tid] = e; __syncthreads();
    for (int s = VV / 2; s > 0; s >>= 1) {
        if (tid < s) red[tid] += red[tid + s];
        __syncthreads();
    }
    w16[base + tid] = (_Float16)(e / red[0]);
}

// ---------------------------------------------------------------------------
// 7) Residual + LayerNorm over D=256
// ---------------------------------------------------------------------------
__global__ void ln_kernel(const float* __restrict__ he, const float* __restrict__ y,
                          const float* __restrict__ lnw, const float* __restrict__ lnb,
                          float* __restrict__ out) {
    __shared__ float red[DD];
    long base = (long)blockIdx.x * DD;
    int tid = threadIdx.x;
    float x = he[base + tid] + y[base + tid];
    red[tid] = x; __syncthreads();
    for (int s = DD / 2; s > 0; s >>= 1) {
        if (tid < s) red[tid] += red[tid + s];
        __syncthreads();
    }
    float mu = red[0] * (1.0f / DD); __syncthreads();
    float d = x - mu;
    red[tid] = d * d; __syncthreads();
    for (int s = DD / 2; s > 0; s >>= 1) {
        if (tid < s) red[tid] += red[tid + s];
        __syncthreads();
    }
    float var = red[0] * (1.0f / DD);
    out[base + tid] = d * rsqrtf(var + 1e-5f) * lnw[tid] + lnb[tid];
}

// ---------------------------------------------------------------------------
// Launcher
// ---------------------------------------------------------------------------
extern "C" void kernel_launch(void* const* d_in, const int* in_sizes, int n_in,
                              void* d_out, int out_size, void* d_ws, size_t ws_size,
                              hipStream_t stream) {
    const float* he   = (const float*)d_in[0];
    const float* df   = (const float*)d_in[1];
    const int*   vi   = (const int*)d_in[2];
    const float* mask = (const float*)d_in[3];
    const float* q_r = (const float*)d_in[4],  *q_i = (const float*)d_in[5];
    const float* q_j = (const float*)d_in[6],  *q_k = (const float*)d_in[7];
    const float* q_b = (const float*)d_in[8];
    const float* k_r = (const float*)d_in[9],  *k_i = (const float*)d_in[10];
    const float* k_j = (const float*)d_in[11], *k_k = (const float*)d_in[12];
    const float* k_b = (const float*)d_in[13];
    const float* v_w = (const float*)d_in[14], *v_b = (const float*)d_in[15];
    const float* o_w = (const float*)d_in[16], *o_b = (const float*)d_in[17];
    const float* g1w = (const float*)d_in[18], *g1b = (const float*)d_in[19];
    const float* g2w = (const float*)d_in[20], *g2b = (const float*)d_in[21];
    const float* lnw = (const float*)d_in[22], *lnb = (const float*)d_in[23];
    float* out = (float*)d_out;

    // ---- workspace carve (256B aligned slices) ----
    char* p = (char*)d_ws;
    auto carve = [&](size_t bytes) {
        void* r = (void*)p;
        p += (bytes + 255) & ~(size_t)255;
        return r;
    };
    _Float16* Wq16  = (_Float16*)carve((size_t)DD * DD * 2);
    _Float16* Wk16  = (_Float16*)carve((size_t)DD * DD * 2);
    _Float16* Wv16  = (_Float16*)carve((size_t)DD * DD * 2);
    _Float16* Wo16  = (_Float16*)carve((size_t)DD * DD * 2);
    _Float16* he16  = (_Float16*)carve((size_t)BB * VV * DD * 2);  // aliased as att16 later
    _Float16* q16   = (_Float16*)carve((size_t)BB * VV * DD * 2);
    _Float16* k16   = (_Float16*)carve((size_t)BB * VV * DD * 2);
    _Float16* vT16  = (_Float16*)carve((size_t)BB * DD * VV * 2);  // [B][D][V]
    float*    sc32  = (float*)   carve((size_t)BB * VV * VV * 4);  // aliased as y32 later
    _Float16* w16   = (_Float16*)carve((size_t)BB * VV * VV * 2);
    float*    vdens = (float*)   carve((size_t)BB * VV * 3 * 4);
    float*    vcnt  = (float*)   carve((size_t)BB * VV * 4);
    float*    gmul  = (float*)   carve((size_t)BB * VV * 4);
    _Float16* att16 = he16;          // he16 dead after v-projection
    float*    y32   = sc32;          // scores dead after softmax

    // ---- gate path ----
    hipMemsetAsync(vdens, 0, (size_t)BB * VV * 3 * 4, stream);
    hipMemsetAsync(vcnt,  0, (size_t)BB * VV * 4, stream);
    {
        long n = (long)BB * NN;
        scatter_kernel<<<(int)((n + 255) / 256), 256, 0, stream>>>(df, vi, mask, vdens, vcnt);
    }
    gate_kernel<<<(BB * VV + 255) / 256, 256, 0, stream>>>(vdens, vcnt, g1w, g1b, g2w, g2b, gmul);

    // ---- f16 weight & activation prep ----
    prep_weights_kernel<<<dim3((DD * DD + 255) / 256, 4), 256, 0, stream>>>(
        q_r, q_i, q_j, q_k, k_r, k_i, k_j, k_k, v_w, o_w, Wq16, Wk16, Wv16, Wo16);
    {
        long n = (long)BB * VV * DD;
        cvt_f16_kernel<<<(int)((n + 255) / 256), 256, 0, stream>>>(he, he16, n);
    }

    // ---- GEMM launch helper: one wave per 32x32 output block ----
    auto blocks_for = [](int M, int N, int batches) {
        long waves = (long)batches * (M / 32) * (N / 32);
        return (int)((waves * 32 + 255) / 256);
    };
    const long sHE = (long)VV * DD;

    // q = he @ Wq^T + q_b   (f16 out)
    wmma_gemm_kernel<0><<<blocks_for(VV, DD, BB), 256, 0, stream>>>(
        he16, sHE, DD, Wq16, 0, DD, q16, sHE, DD, q_b, nullptr, 0, 1.0f, VV, DD, DD, BB);
    // k = he @ Wk^T + k_b
    wmma_gemm_kernel<0><<<blocks_for(VV, DD, BB), 256, 0, stream>>>(
        he16, sHE, DD, Wk16, 0, DD, k16, sHE, DD, k_b, nullptr, 0, 1.0f, VV, DD, DD, BB);
    // v^T = (he @ v_w^T + v_b)^T  -> [B][D][V] so weights@v has contiguous B-operand
    wmma_gemm_kernel<1><<<blocks_for(VV, DD, BB), 256, 0, stream>>>(
        he16, sHE, DD, Wv16, 0, DD, vT16, (long)DD * VV, VV, v_b, nullptr, 0, 1.0f, VV, DD, DD, BB);
    // scores = (q @ k^T) / sqrt(Q) * (1 + 0.5*gate[col])   (f32 out, fused epilogue)
    wmma_gemm_kernel<2><<<blocks_for(VV, VV, BB), 256, 0, stream>>>(
        q16, sHE, DD, k16, sHE, DD, sc32, (long)VV * VV, VV,
        nullptr, gmul, VV, 0.125f, VV, VV, DD, BB);
    // softmax -> f16 weights
    softmax_kernel<<<BB * VV, VV, 0, stream>>>(sc32, w16);
    // att = weights @ v      (B-operand = rows of v^T)
    wmma_gemm_kernel<0><<<blocks_for(VV, DD, BB), 256, 0, stream>>>(
        w16, (long)VV * VV, VV, vT16, (long)DD * VV, VV, att16, sHE, DD,
        nullptr, nullptr, 0, 1.0f, VV, DD, VV, BB);
    // y = att @ o_w^T + o_b  (f32 out)
    wmma_gemm_kernel<2><<<blocks_for(VV, DD, BB), 256, 0, stream>>>(
        att16, sHE, DD, Wo16, 0, DD, y32, sHE, DD, o_b, nullptr, 0, 1.0f, VV, DD, DD, BB);
    // residual + layernorm
    ln_kernel<<<BB * VV, DD, 0, stream>>>(he, y32, lnw, lnb, out);
}
